// VectorQuantizer_35845797052743
// MI455X (gfx1250) — compile-verified, matrix operand order checked by reading, not run on hardware
//
#include <hip/hip_runtime.h>

#define N_EMB   1024
#define EMB_DIM 256
#define HW      4096
#define BATCH   32
#define M_ROWS  (BATCH * HW)           // 131072 query rows
#define TOT     (BATCH * EMB_DIM * HW) // 33554432 elements of x

typedef __bf16        v16bf __attribute__((ext_vector_type(16)));
typedef float         v8f   __attribute__((ext_vector_type(8)));
typedef unsigned int  u32x4 __attribute__((ext_vector_type(4)));
typedef int           i32x4 __attribute__((ext_vector_type(4)));
typedef float         f32x4 __attribute__((ext_vector_type(4)));

union Frag {
    v16bf v;
    u32x4 q[2];
};

// CDNA5 async global->LDS path (ASYNCcnt) if the toolchain declares it.
#if defined(__has_builtin)
# if __has_builtin(__builtin_amdgcn_global_load_async_to_lds_b128)
#  define VQ_ASYNC 1
# endif
#endif
#ifndef VQ_ASYNC
# define VQ_ASYNC 0
#endif

__device__ __forceinline__ void vq_wait_async() {
#if VQ_ASYNC
# if defined(__has_builtin) && __has_builtin(__builtin_amdgcn_s_wait_asynccnt)
    __builtin_amdgcn_s_wait_asynccnt(0);
# else
    asm volatile("s_wait_asynccnt 0x0" ::: "memory");
# endif
#endif
}

__device__ __forceinline__ unsigned short f32_to_bf16(float f) {
    unsigned int u = __float_as_uint(f);
    unsigned int lsb = (u >> 16) & 1u;
    u += 0x7FFFu + lsb;              // round-to-nearest-even
    return (unsigned short)(u >> 16);
}

// ---------------------------------------------------------------------------
// Kernel 1: codebook f32 -> bf16, and e2[j] = ||emb_j||^2
// ---------------------------------------------------------------------------
__global__ __launch_bounds__(EMB_DIM)
void vq_prep(const float* __restrict__ emb_w,
             unsigned short* __restrict__ emb_bf,
             float* __restrict__ e2) {
    __shared__ float red[EMB_DIM];
    const int row = blockIdx.x;
    const int t   = threadIdx.x;
    float v = emb_w[row * EMB_DIM + t];
    emb_bf[row * EMB_DIM + t] = f32_to_bf16(v);
    red[t] = v * v;
    __syncthreads();
    for (int s = EMB_DIM / 2; s > 0; s >>= 1) {
        if (t < s) red[t] += red[t + s];
        __syncthreads();
    }
    if (t == 0) e2[row] = red[0];
}

// ---------------------------------------------------------------------------
// Kernel 2: bf16 WMMA distance GEMM + fused argmin over 1024 codes.
//   argmin_j (e2[j] - 2 * <x_row, emb_j>)   (x^2 term constant per row)
// 256 threads = 8 waves; each wave owns a 16-row strip of a 128-row tile.
// Codebook staged into LDS in 64-code chunks, double-buffered with
// CDNA5 async global->LDS loads (ASYNCcnt) when available.
// ---------------------------------------------------------------------------
#define SROW   264   // LDS row stride in bf16 (256 + pad, 16B-aligned rows)
#define BCHUNK 64    // codes per LDS B chunk
#define NCHUNK (N_EMB / BCHUNK)   // 16 chunks

__device__ __forceinline__ void vq_issue_chunk(const unsigned short* __restrict__ emb_bf,
                                               int jbase,
                                               unsigned short* dst,   // LDS (generic)
                                               int t) {
    // 64 codes x 256 K bf16 = 2048 x 16B granules; 256 threads -> 8 each.
#pragma unroll
    for (int it = 0; it < 8; ++it) {
        const int g  = it * 256 + t;
        const int r  = g >> 5;       // code row in chunk 0..63
        const int kg = g & 31;       // 16B granule along K
        const unsigned short* gp = emb_bf + (size_t)(jbase + r) * EMB_DIM + kg * 8;
        unsigned short*       lp = dst + r * SROW + kg * 8;
#if VQ_ASYNC
        __builtin_amdgcn_global_load_async_to_lds_b128(
            (__attribute__((address_space(1))) i32x4*)gp,
            (__attribute__((address_space(3))) i32x4*)lp,
            0, 0);
#else
        *(u32x4*)lp = *(const u32x4*)gp;
#endif
    }
}

__global__ __launch_bounds__(256)
void vq_argmin(const float* __restrict__ x,
               const unsigned short* __restrict__ emb_bf,
               const float* __restrict__ e2,
               int* __restrict__ idx) {
    __shared__ unsigned short smA[128 * SROW];          // 66 KB: 128 rows x 256 K
    __shared__ unsigned short smB[2][BCHUNK * SROW];    // 2 x 33 KB double buffer

    const int t       = threadIdx.x;
    const int lid     = t & 31;
    const int w       = t >> 5;                 // wave id 0..7
    const int rowBase = blockIdx.x * 128;       // global query-row base
    const int bb      = rowBase / HW;           // batch index (tile never crosses batch)
    const int n0      = rowBase % HW;           // spatial offset

    // Kick off async staging of B chunk 0 while we convert the A tile.
    vq_issue_chunk(emb_bf, 0, &smB[0][0], t);

    // Stage 128(n) x 256(c) tile of x into LDS as bf16, [n][c] layout.
    // x is (b, c, h*w): contiguous along n for fixed c -> coalesced float4 loads.
    for (int it = 0; it < 32; ++it) {
        const int flat = it * 256 + t;
        const int c    = flat >> 5;             // 0..255
        const int nch  = flat & 31;             // float4 chunk along n
        f32x4 xv = *(const f32x4*)(x + (size_t)(bb * EMB_DIM + c) * HW + n0 + nch * 4);
#pragma unroll
        for (int k = 0; k < 4; ++k)
            smA[(nch * 4 + k) * SROW + c] = f32_to_bf16(xv[k]);
    }
    __syncthreads();

    // A fragments resident in registers: 8 frags cover K = 0..255.
    // 16-bit A 16x32 layout: lane L -> row L%16, Kbase = (L/16)*8,
    // elems[0..7] = K kb..kb+7, elems[8..15] = K kb+16..kb+23.
    const int r     = lid & 15;
    const int halfk = (lid >> 4) * 8;
    Frag a[8];
#pragma unroll
    for (int kf = 0; kf < 8; ++kf) {
        const unsigned short* p = &smA[(w * 16 + r) * SROW + kf * 32 + halfk];
        a[kf].q[0] = *(const u32x4*)p;
        a[kf].q[1] = *(const u32x4*)(p + 16);
    }

    float runv[8];
    int   runi[8];
#pragma unroll
    for (int v = 0; v < 8; ++v) { runv[v] = 3.0e38f; runi[v] = 0; }

    const int ncol = lid & 15;                  // D-matrix: lane -> column

    for (int cc = 0; cc < NCHUNK; ++cc) {
        // Chunk cc has landed; all waves are past compute(cc-1), so the
        // opposite buffer is free to overwrite.
        vq_wait_async();
        __syncthreads();
        if (cc + 1 < NCHUNK)
            vq_issue_chunk(emb_bf, (cc + 1) * BCHUNK, &smB[(cc + 1) & 1][0], t);

        const unsigned short* bbase = &smB[cc & 1][0];
        for (int sub = 0; sub < 4; ++sub) {
            const int j = cc * 4 + sub;         // 16-code group index (0..63)
            const unsigned short* bp = bbase + (sub * 16 + r) * SROW + halfk;
            v8f acc = {0.f, 0.f, 0.f, 0.f, 0.f, 0.f, 0.f, 0.f};
#pragma unroll
            for (int kf = 0; kf < 8; ++kf) {
                Frag b;
                b.q[0] = *(const u32x4*)(bp + kf * 32);
                b.q[1] = *(const u32x4*)(bp + kf * 32 + 16);
                acc = __builtin_amdgcn_wmma_f32_16x16x32_bf16(
                    false, a[kf].v, false, b.v, (short)0, acc, false, false);
            }
            const float e2v = e2[j * 16 + ncol];
            const int   ni  = j * 16 + ncol;
#pragma unroll
            for (int v = 0; v < 8; ++v) {
                float dv = e2v - 2.0f * acc[v];
                bool better = (dv < runv[v]) || (dv == runv[v] && ni < runi[v]);
                runv[v] = better ? dv : runv[v];
                runi[v] = better ? ni : runi[v];
            }
        }
    }

    // Cross-lane argmin within each 16-lane half (lanes share row m = v + 8*(L/16)).
#pragma unroll
    for (int v = 0; v < 8; ++v) {
        float rv = runv[v];
        int   ri = runi[v];
#pragma unroll
        for (int m = 1; m < 16; m <<= 1) {
            float ov = __shfl_xor(rv, m, 32);
            int   oi = __shfl_xor(ri, m, 32);
            bool take = (ov < rv) || (ov == rv && oi < ri);
            rv = take ? ov : rv;
            ri = take ? oi : ri;
        }
        if ((lid & 15) == 0)
            idx[rowBase + w * 16 + (lid >> 4) * 8 + v] = ri;
    }
}

// ---------------------------------------------------------------------------
// Kernel 3: out = x (float4 copy) fused with per-block loss partial:
//   sum (x - emb_w[idx[row]])^2 in f32 from original data.
// ---------------------------------------------------------------------------
__global__ __launch_bounds__(256)
void vq_loss_copy(const float* __restrict__ x,
                  const float* __restrict__ emb_w,
                  const int* __restrict__ idx,
                  float* __restrict__ out,
                  float* __restrict__ partial) {
    __shared__ float red[256];
    const int t   = threadIdx.x;
    const int gid = blockIdx.x * 256 + t;       // float4 index
    f32x4 xv = ((const f32x4*)x)[gid];
    ((f32x4*)out)[gid] = xv;

    const int i   = gid * 4;
    const int b   = i / (EMB_DIM * HW);
    const int rem = i - b * (EMB_DIM * HW);
    const int c   = rem / HW;
    const int n   = rem - c * HW;
    const int rb  = b * HW + n;                 // 4 consecutive rows

    const int4 id4 = ((const int4*)idx)[rb >> 2];
    const int ids[4] = { id4.x, id4.y, id4.z, id4.w };

    float s = 0.f;
#pragma unroll
    for (int k = 0; k < 4; ++k) {
        float e = emb_w[(size_t)ids[k] * EMB_DIM + c];
        float d = xv[k] - e;
        s += d * d;
    }
    red[t] = s;
    __syncthreads();
    for (int st = 128; st > 0; st >>= 1) {
        if (t < st) red[t] += red[t + st];
        __syncthreads();
    }
    if (t == 0) partial[blockIdx.x] = red[0];
}

// ---------------------------------------------------------------------------
// Kernel 4: deterministic final reduction; loss = 1.25 * mean(diff^2)
// ---------------------------------------------------------------------------
__global__ __launch_bounds__(256)
void vq_finalize(const float* __restrict__ partial, int nPart,
                 float* __restrict__ out_loss) {
    __shared__ float red[256];
    const int t = threadIdx.x;
    float s = 0.f;
    for (int i = t; i < nPart; i += 256) s += partial[i];
    red[t] = s;
    __syncthreads();
    for (int st = 128; st > 0; st >>= 1) {
        if (t < st) red[t] += red[t + st];
        __syncthreads();
    }
    if (t == 0) *out_loss = 1.25f * red[0] / (float)TOT;
}

// ---------------------------------------------------------------------------
extern "C" void kernel_launch(void* const* d_in, const int* in_sizes, int n_in,
                              void* d_out, int out_size, void* d_ws, size_t ws_size,
                              hipStream_t stream) {
    const float* x     = (const float*)d_in[0];   // (32, 256, 64, 64) f32
    const float* emb_w = (const float*)d_in[1];   // (1024, 256) f32
    float* out = (float*)d_out;                   // [x (TOT floats)] + [loss]

    char* ws = (char*)d_ws;
    unsigned short* emb_bf = (unsigned short*)ws;             // 512 KB bf16 codebook
    float* e2      = (float*)(ws + 524288);                   // 4 KB
    int*   idx     = (int*)  (ws + 528384);                   // 512 KB (131072 ints)
    float* partial = (float*)(ws + 1052672);                  // 128 KB partials

    vq_prep<<<N_EMB, EMB_DIM, 0, stream>>>(emb_w, emb_bf, e2);
    vq_argmin<<<M_ROWS / 128, 256, 0, stream>>>(x, emb_bf, e2, idx);

    const int nPart = TOT / (256 * 4);   // 32768 blocks
    vq_loss_copy<<<nPart, 256, 0, stream>>>(x, emb_w, idx, out, partial);
    vq_finalize<<<1, 256, 0, stream>>>(partial, nPart, out + TOT);
}
